// myAttention_47433618817438
// MI455X (gfx1250) — compile-verified, matrix operand order checked by reading, not run on hardware
//
#include <hip/hip_runtime.h>
#include <hip/hip_bf16.h>

// Problem sizes (fixed by the reference)
#define Bsz 64
#define Csz 256
#define Tsz 4096
#define Osz 256
#define NT  64          // time-tile per workgroup
#define C_PAD 264       // bf16 elements per LDS row (528B, conflict-free)

typedef __bf16       v16bf __attribute__((ext_vector_type(16)));
typedef float        v8f   __attribute__((ext_vector_type(8)));
typedef unsigned int u32x4 __attribute__((ext_vector_type(4)));

union FragU {
  v16bf v;
  u32x4 q[2];
};

__device__ __forceinline__ unsigned short f32_bf16(float f) {
  union { float f; unsigned u; } x{f};
  unsigned u = x.u;
  return (unsigned short)((u + 0x7FFFu + ((u >> 16) & 1u)) >> 16);  // RNE
}

// fast tanh: v_tanh_f32 on the TRANS pipe if the builtin exists
#if __has_builtin(__builtin_amdgcn_tanhf)
__device__ __forceinline__ float fast_tanh(float x) { return __builtin_amdgcn_tanhf(x); }
#elif __has_builtin(__builtin_amdgcn_tanh_f32)
__device__ __forceinline__ float fast_tanh(float x) { return __builtin_amdgcn_tanh_f32(x); }
#else
__device__ __forceinline__ float fast_tanh(float x) { return tanhf(x); }
#endif

__device__ __forceinline__ float fast_sigmoid(float x) {
#if __has_builtin(__builtin_amdgcn_rcpf)
  return __builtin_amdgcn_rcpf(1.0f + __expf(-x));   // v_exp + v_rcp (TRANS)
#else
  return 1.0f / (1.0f + __expf(-x));
#endif
}

__device__ __forceinline__ unsigned pack_f16x2(float a, float b) {
#if __has_builtin(__builtin_amdgcn_cvt_pkrtz)
  auto h2 = __builtin_amdgcn_cvt_pkrtz(a, b);        // v_cvt_pk_rtz_f16_f32
  unsigned u;
  __builtin_memcpy(&u, &h2, 4);
  return u;
#else
  union { _Float16 h[2]; unsigned u; } x;
  x.h[0] = (_Float16)a;
  x.h[1] = (_Float16)b;
  return x.u;
#endif
}

__device__ __forceinline__ float2 unpack_f16x2(unsigned u) {
  union { unsigned u; _Float16 h[2]; } x{u};
  return make_float2((float)x.h[0], (float)x.h[1]);
}

// A-fragment: pre-swizzled weights, one contiguous 32B chunk per lane.
__device__ __forceinline__ v16bf load_a_frag(const unsigned short* __restrict__ Wf,
                                             int mtile, int k8) {
  const int lane = threadIdx.x & 31;
  const unsigned short* p = Wf + (((mtile * 8 + k8) * 32 + lane) << 4);
  FragU f;
  f.q[0] = *(const u32x4*)p;
  f.q[1] = *(const u32x4*)(p + 8);
  return f.v;
}

// B-fragment from LDS tile laid out [t][c] (c contiguous), bf16.
__device__ __forceinline__ v16bf load_b_frag(const unsigned short* __restrict__ tile,
                                             int n4, int k8) {
  const int lane = threadIdx.x & 31;
  const int n    = n4 * 16 + (lane & 15);
  const int klo  = k8 * 32 + ((lane & 16) ? 8 : 0);
  const unsigned short* p = tile + n * C_PAD + klo;
  FragU f;
  f.q[0] = *(const u32x4*)p;        // K = klo .. klo+7
  f.q[1] = *(const u32x4*)(p + 16); // K = klo+16 .. klo+23
  return f.v;
}

// ---------------------------------------------------------------------------
// Kernel 0: convert the 4 weight matrices (f32 row-major [256][256]) to bf16
// in A-fragment order: ((mtile*8 + k8)*32 + lane)*16 + j
// ---------------------------------------------------------------------------
__global__ void __launch_bounds__(256) frag_weights_kernel(
    const float* __restrict__ Wt, const float* __restrict__ Ws,
    const float* __restrict__ Ww, const float* __restrict__ Wfin,
    unsigned short* __restrict__ out) {
  const int g = blockIdx.x * 256 + threadIdx.x;   // 4 * 65536 total
  const int mat = g >> 16;
  const int d   = g & 65535;
  const int j     = d & 15;
  const int lane  = (d >> 4) & 31;
  const int k8    = (d >> 9) & 7;
  const int mtile = (d >> 12) & 15;
  const int M   = mtile * 16 + (lane & 15);
  const int col = k8 * 32 + ((lane & 16) ? 8 : 0) + ((j < 8) ? j : j + 8);
  const float* W = (mat == 0) ? Wt : (mat == 1) ? Ws : (mat == 2) ? Ww : Wfin;
  out[g] = f32_bf16(W[M * Csz + col]);
}

// ---------------------------------------------------------------------------
// Packed-f16 epilogue store: results go into the FIRST 2048 floats of each
// row's own 4096-float output slot as f16x2 (t, t+1), halving store traffic.
// lane pairs exchange their t-values with shfl_xor; even lanes store r=0..3,
// odd lanes store r=4..7, so all 32 lanes issue b32 stores.
// ---------------------------------------------------------------------------
__device__ __forceinline__ void store_rows_packed(
    const v8f acc[2][4], const float* __restrict__ bias,
    float* __restrict__ buf, int b, int wave, int hi8, int nloc, int tbase) {
  unsigned* ub = (unsigned*)buf;
  const bool evenl = (nloc & 1) == 0;
#pragma unroll
  for (int m = 0; m < 2; ++m) {
    const int o0 = (wave * 2 + m) * 16 + hi8;
    float bv[8];
#pragma unroll
    for (int r = 0; r < 8; ++r) bv[r] = bias[o0 + r];
#pragma unroll
    for (int n4 = 0; n4 < 4; ++n4) {
      const int t  = tbase + n4 * 16 + nloc;
      const int tp = t >> 1;                     // f16-pair index within row
      unsigned pk[8];
#pragma unroll
      for (int r = 0; r < 8; ++r) {
        float v = acc[m][n4][r] + bv[r];
        float w = __shfl_xor(v, 1);              // partner t value
        pk[r] = evenl ? pack_f16x2(v, w) : pack_f16x2(w, v);
      }
      const int r0 = evenl ? 0 : 4;
#pragma unroll
      for (int rr = 0; rr < 4; ++rr) {
        const int r = r0 + rr;
        ub[(((size_t)b << 8) + (o0 + r)) * Tsz + tp] = pk[r];
      }
    }
  }
}

// ---------------------------------------------------------------------------
// Kernel 1: fused 4-GEMM tile kernel.
//   z = W_w @ (tanh(W_t x + b_t) * sigm(W_s x + b_s)) + b_w  -> zbuf (packed f16)
//   f = W_final @ x + b_final                                -> fbuf (packed f16)
// ---------------------------------------------------------------------------
__global__ void __launch_bounds__(256) fused_gemm_kernel(
    const float* __restrict__ X, const unsigned short* __restrict__ Wfrag,
    const float* __restrict__ bt, const float* __restrict__ bs,
    const float* __restrict__ bw, const float* __restrict__ bfin,
    float* __restrict__ zbuf, float* __restrict__ fbuf) {
  __shared__ unsigned short xs[NT * C_PAD];  // x tile, bf16, [t][c]
  __shared__ unsigned short hs[NT * C_PAD];  // h tile, bf16, [t][c]

  const int b     = blockIdx.x >> 6;
  const int tbase = (blockIdx.x & 63) * NT;
  const int tid   = threadIdx.x;
  const int wave  = tid >> 5;
  const int lane  = tid & 31;

  // ---- stage x tile -> LDS (bf16, transposed to [t][c]) ----
  {
    const int c = tid;
    const float4* xp = (const float4*)(X + ((size_t)b * Csz + c) * Tsz + tbase);
#pragma unroll
    for (int j4 = 0; j4 < 16; ++j4) {
      float4 v = xp[j4];
      const int t0 = j4 * 4;
      xs[(t0 + 0) * C_PAD + c] = f32_bf16(v.x);
      xs[(t0 + 1) * C_PAD + c] = f32_bf16(v.y);
      xs[(t0 + 2) * C_PAD + c] = f32_bf16(v.z);
      xs[(t0 + 3) * C_PAD + c] = f32_bf16(v.w);
    }
  }
  __syncthreads();

  const unsigned short* Wt = Wfrag;
  const unsigned short* Ws = Wfrag + 65536;
  const unsigned short* Ww = Wfrag + 131072;
  const unsigned short* Wf = Wfrag + 196608;

  const int hi8  = (lane & 16) ? 8 : 0;
  const int nloc = lane & 15;

  // ---- stage A: tanh & sigm GEMMs (share B fragments) ----
  v8f acc_t[2][4] = {};
  v8f acc_s[2][4] = {};
  for (int k8 = 0; k8 < 8; ++k8) {
    __builtin_prefetch(Wt + (((wave * 2 * 8 + ((k8 + 1) & 7)) * 32) << 4), 0, 0);
    v16bf at[2], asg[2];
#pragma unroll
    for (int m = 0; m < 2; ++m) {
      at[m]  = load_a_frag(Wt, wave * 2 + m, k8);
      asg[m] = load_a_frag(Ws, wave * 2 + m, k8);
    }
#pragma unroll
    for (int n4 = 0; n4 < 4; ++n4) {
      v16bf bfr = load_b_frag(xs, n4, k8);
#pragma unroll
      for (int m = 0; m < 2; ++m) {
        acc_t[m][n4] = __builtin_amdgcn_wmma_f32_16x16x32_bf16(
            false, at[m],  false, bfr, (short)0, acc_t[m][n4], false, false);
        acc_s[m][n4] = __builtin_amdgcn_wmma_f32_16x16x32_bf16(
            false, asg[m], false, bfr, (short)0, acc_s[m][n4], false, false);
      }
    }
  }

  // ---- epilogue A: h = tanh(.)*sigmoid(.) -> hs (bf16 [t][c]) ----
#pragma unroll
  for (int m = 0; m < 2; ++m) {
    const int c0 = (wave * 2 + m) * 16 + hi8;
    float btv[8], bsv[8];
#pragma unroll
    for (int r = 0; r < 8; ++r) { btv[r] = bt[c0 + r]; bsv[r] = bs[c0 + r]; }
#pragma unroll
    for (int n4 = 0; n4 < 4; ++n4) {
      unsigned pk[4];
#pragma unroll
      for (int r2 = 0; r2 < 4; ++r2) {
        const int r0 = r2 * 2, r1 = r2 * 2 + 1;
        float g0 = fast_tanh(acc_t[m][n4][r0] + btv[r0]) *
                   fast_sigmoid(acc_s[m][n4][r0] + bsv[r0]);
        float g1 = fast_tanh(acc_t[m][n4][r1] + btv[r1]) *
                   fast_sigmoid(acc_s[m][n4][r1] + bsv[r1]);
        pk[r2] = (unsigned)f32_bf16(g0) | ((unsigned)f32_bf16(g1) << 16);
      }
      const int n = n4 * 16 + nloc;
      u32x4 wv = {pk[0], pk[1], pk[2], pk[3]};
      *(u32x4*)(hs + n * C_PAD + c0) = wv;
    }
  }

  // ---- f GEMM: W_final @ x (uses xs; before barrier) ----
  v8f acc_f[2][4] = {};
  for (int k8 = 0; k8 < 8; ++k8) {
    v16bf af[2];
#pragma unroll
    for (int m = 0; m < 2; ++m) af[m] = load_a_frag(Wf, wave * 2 + m, k8);
#pragma unroll
    for (int n4 = 0; n4 < 4; ++n4) {
      v16bf bfr = load_b_frag(xs, n4, k8);
#pragma unroll
      for (int m = 0; m < 2; ++m)
        acc_f[m][n4] = __builtin_amdgcn_wmma_f32_16x16x32_bf16(
            false, af[m], false, bfr, (short)0, acc_f[m][n4], false, false);
    }
  }
  store_rows_packed(acc_f, bfin, fbuf, b, wave, hi8, nloc, tbase);

  __syncthreads();  // hs fully written by all waves

  // ---- z GEMM: W_w @ h (uses hs) ----
  v8f acc_z[2][4] = {};
  for (int k8 = 0; k8 < 8; ++k8) {
    v16bf aw[2];
#pragma unroll
    for (int m = 0; m < 2; ++m) aw[m] = load_a_frag(Ww, wave * 2 + m, k8);
#pragma unroll
    for (int n4 = 0; n4 < 4; ++n4) {
      v16bf bfr = load_b_frag(hs, n4, k8);
#pragma unroll
      for (int m = 0; m < 2; ++m)
        acc_z[m][n4] = __builtin_amdgcn_wmma_f32_16x16x32_bf16(
            false, aw[m], false, bfr, (short)0, acc_z[m][n4], false, false);
    }
  }
  store_rows_packed(acc_z, bw, zbuf, b, wave, hi8, nloc, tbase);
}

// ---------------------------------------------------------------------------
// Kernel 2: per-(b,o) masked softmax over T + weighted reduction.
//   zbuf row: in packed-f16 z (first 2048 floats), out f32 a (full row)
//   fbuf row: in packed-f16 f,                     out f32 a2 = f*a
//   out[row] = sum_t a2
// ---------------------------------------------------------------------------
__global__ void __launch_bounds__(256) row_softmax_kernel(
    const int* __restrict__ lens, const int* __restrict__ levels,
    float* __restrict__ zbuf, float* __restrict__ fbuf,
    float* __restrict__ out) {
  const int row = blockIdx.x;           // row = b*O + o
  const int b   = row >> 8;             // O == 256
  const int tid = threadIdx.x;
  const int k   = lens[b] >> (levels[0] - 1);   // valid length, >= 1

  float* zr = zbuf + (size_t)row * Tsz;
  float* fr = fbuf + (size_t)row * Tsz;
  const unsigned* zp = (const unsigned*)zr;     // packed f16x2, [0,2048)
  const unsigned* fp = (const unsigned*)fr;

  __shared__ float red[256];

  // phase 1: load ALL packed z and f values (reads complete before any write)
  float zv[16], fv[16];
  float m = -INFINITY;
#pragma unroll
  for (int j = 0; j < 8; ++j) {
    const int p  = tid + j * 256;       // pair index; t = 2p, 2p+1
    float2 z2 = unpack_f16x2(zp[p]);
    float2 f2 = unpack_f16x2(fp[p]);
    zv[2 * j] = z2.x;  zv[2 * j + 1] = z2.y;
    fv[2 * j] = f2.x;  fv[2 * j + 1] = f2.y;
    const int t0 = 2 * p;
    if (t0 < k)     m = fmaxf(m, z2.x);
    if (t0 + 1 < k) m = fmaxf(m, z2.y);
  }
  red[tid] = m;
  __syncthreads();
  for (int s = 128; s > 0; s >>= 1) {
    if (tid < s) red[tid] = fmaxf(red[tid], red[tid + s]);
    __syncthreads();
  }
  m = red[0];
  __syncthreads();

  // phase 2: exp + sum
  float ssum = 0.0f;
#pragma unroll
  for (int j = 0; j < 8; ++j) {
    const int t0 = 2 * (tid + j * 256);
    float e0 = (t0 < k)     ? __expf(zv[2 * j]     - m) : 0.0f;
    float e1 = (t0 + 1 < k) ? __expf(zv[2 * j + 1] - m) : 0.0f;
    zv[2 * j] = e0;  zv[2 * j + 1] = e1;
    ssum += e0 + e1;
  }
  red[tid] = ssum;
  __syncthreads();
  for (int s = 128; s > 0; s >>= 1) {
    if (tid < s) red[tid] += red[tid + s];
    __syncthreads();
  }
  const float inv = 1.0f / red[0];
  __syncthreads();   // everyone has inv; all packed reads long done

  // phase 3: write a (f32), a2 = f*a (f32), accumulate output
  float acc = 0.0f;
#pragma unroll
  for (int j = 0; j < 8; ++j) {
    const int t0 = 2 * (tid + j * 256);
    float a0 = zv[2 * j] * inv;
    float a1 = zv[2 * j + 1] * inv;
    float q0 = fv[2 * j] * a0;
    float q1 = fv[2 * j + 1] * a1;
    *(float2*)(zr + t0) = make_float2(a0, a1);   // a
    *(float2*)(fr + t0) = make_float2(q0, q1);   // a2
    acc += q0 + q1;
  }
  red[tid] = acc;
  __syncthreads();
  for (int s = 128; s > 0; s >>= 1) {
    if (tid < s) red[tid] += red[tid + s];
    __syncthreads();
  }
  if (tid == 0) out[row] = red[0];
}

// ---------------------------------------------------------------------------
extern "C" void kernel_launch(void* const* d_in, const int* in_sizes, int n_in,
                              void* d_out, int out_size, void* d_ws, size_t ws_size,
                              hipStream_t stream) {
  const float* X     = (const float*)d_in[0];
  const int*   lens  = (const int*)d_in[1];
  const int*   lvls  = (const int*)d_in[2];
  const float* Wt    = (const float*)d_in[3];
  const float* bt    = (const float*)d_in[4];
  const float* Ws    = (const float*)d_in[5];
  const float* bs    = (const float*)d_in[6];
  const float* Ww    = (const float*)d_in[7];
  const float* bw    = (const float*)d_in[8];
  const float* Wfin  = (const float*)d_in[9];
  const float* bfin  = (const float*)d_in[10];

  float* out   = (float*)d_out;                           // [B*O]
  float* abuf  = out + (size_t)Bsz * Osz;                 // [B*O*T] (packed z -> a)
  float* a2buf = abuf + (size_t)Bsz * Osz * Tsz;          // [B*O*T] (packed f -> a2)

  unsigned short* wfrag = (unsigned short*)d_ws;          // 4*65536 bf16 = 512KB

  frag_weights_kernel<<<1024, 256, 0, stream>>>(Wt, Ws, Ww, Wfin, wfrag);

  fused_gemm_kernel<<<Bsz * (Tsz / NT), 256, 0, stream>>>(
      X, wfrag, bt, bs, bw, bfin, abuf, a2buf);

  row_softmax_kernel<<<Bsz * Osz, 256, 0, stream>>>(lens, lvls, abuf, a2buf, out);
}
